// _PairwiseMessages_80161269612628
// MI455X (gfx1250) — matrix operand order repaired and checked by reading, not compiled, not run on hardware
//
#include <hip/hip_runtime.h>
#include <hip/hip_bf16.h>

typedef __attribute__((ext_vector_type(16))) _Float16 v16h;
typedef __attribute__((ext_vector_type(8)))  float    v8f;

#define BB 4
#define NQ 384
#define NK 384

// ---------------------------------------------------------------------------
// Kernel 1: qh[b,q,f] = (q_inv@Wq + bq) @ W1[0:64,:] + b1      (f32, [1536,128])
//           kh[b,k,f] = (k_inv@Wk + bk) @ W1[64:128,:]         (f32, [1536,128])
// Tiny (~150 MFLOP total) — plain VALU kernel.
// ---------------------------------------------------------------------------
__global__ __launch_bounds__(128) void precompute_qh_kh(
    const float* __restrict__ q_inv, const float* __restrict__ k_inv,
    const float* __restrict__ Wq, const float* __restrict__ bq,
    const float* __restrict__ Wk, const float* __restrict__ bk,
    const float* __restrict__ W1, const float* __restrict__ b1,
    float* __restrict__ qh, float* __restrict__ kh)
{
  const int tid = threadIdx.x;
  const int gb  = blockIdx.x;              // 0 .. 2*B*N-1
  const int row = gb % (BB * NQ);
  const bool isK = gb >= (BB * NQ);

  __shared__ float inv_s[256];
  __shared__ float msg_s[64];

  const float* src = isK ? k_inv : q_inv;
  inv_s[tid]       = src[row * 256 + tid];
  inv_s[tid + 128] = src[row * 256 + tid + 128];
  __syncthreads();

  if (tid < 64) {
    const float* W = isK ? Wk : Wq;
    float s = isK ? bk[tid] : bq[tid];
    for (int i = 0; i < 256; ++i) s += inv_s[i] * W[i * 64 + tid];
    msg_s[tid] = s;
  }
  __syncthreads();

  float s = isK ? 0.f : b1[tid];
  const float* W1p = W1 + (isK ? 64 : 0) * 128;
  for (int d = 0; d < 64; ++d) s += msg_s[d] * W1p[d * 128 + tid];
  (isK ? kh : qh)[row * 128 + tid] = s;
}

// ---------------------------------------------------------------------------
// Kernel 2: fused pairwise kernel. Workgroup = 128 thr (4 waves).
// One WG owns (b, 16-q block, 64-k tile); waves own 16 k each and loop over
// the 16 q values. ke / kh / W-fragments staged once per WG; qe/qh per q.
// Per (wave, q):
//   D[16,64]  = per-channel dotprods (VALU, LDS float2 reads)
//   H[16,128] = SiLU(D @ W1c + qh[q] + kh[k])   (16x v_wmma_f32_16x16x32_f16)
//   O[16,64]  = H @ W2 + b2                     (16x v_wmma_f32_16x16x32_f16)
// ---------------------------------------------------------------------------
__global__ __launch_bounds__(128) void fused_pairwise(
    const float* __restrict__ q_equi, const float* __restrict__ k_equi,
    const float* __restrict__ qh, const float* __restrict__ kh,
    const float* __restrict__ W1, const float* __restrict__ W2,
    const float* __restrict__ b2, float* __restrict__ out)
{
  __shared__ float ke_s[3][64][68];     // [c][k][d] pad-68 (float2-friendly, bank-safe)
  __shared__ float kh_s[64][132];       // [k][f] pad-132 (bank-safe across half-waves)
  __shared__ float qe_s[3][64];
  __shared__ float qh_s[128];
  __shared__ float b2_s[64];
  __shared__ __align__(32) _Float16 w1f[2][8][32][16];  // [kstep][ntile][lane][i] B-frags
  __shared__ __align__(32) _Float16 w2f[4][4][32][16];  // [kstep][ntile][lane][i]
  __shared__ __align__(32) _Float16 h_pack[4][4][32][16]; // per-wave A2 frags, packed

  const int tid   = threadIdx.x;
  const int blk   = blockIdx.x;
  const int kt    = blk % 6;                    // 6 k-tiles of 64
  const int qb    = (blk / 6) % (NQ / 16);      // 24 q-blocks of 16
  const int b     = blk / (6 * (NQ / 16));
  const int kbase = kt * 64;
  const int qbase = qb * 16;

  // ---- one-time staging: k_equi tile ([c][k][d], coalesced over d) ----
  for (int idx = tid; idx < 64 * 3 * 64; idx += 128) {
    int d  = idx & 63;
    int c  = (idx >> 6) % 3;
    int kl = idx / 192;
    ke_s[c][kl][d] = k_equi[(((b * NK + kbase + kl) * 3 + c) << 6) + d];
  }
  // ---- one-time staging: kh tile (vectorized float4, rows pad-132 stay 16B-aligned) ----
  for (int idx = tid; idx < 64 * 32; idx += 128) {   // 2048 float4's
    int kl = idx >> 5;
    int fv = (idx & 31) << 2;
    const float4 v = *(const float4*)&kh[(b * NK + kbase + kl) * 128 + fv];
    *(float4*)&kh_s[kl][fv] = v;
  }
  if (tid < 64) b2_s[tid] = b2[tid];

  // ---- one-time: pack W1c = W1[128:192,:] and W2 into f16 B-fragment order ----
  // B frag (32x16), lane n=lane&15, hf=lane>>4: b[i] = B[K = 32*s + 8*hf + (i<8?i:i+8), n]
  for (int idx = tid; idx < 2 * 8 * 32 * 16; idx += 128) {
    int i = idx & 15, lane = (idx >> 4) & 31, t = (idx >> 9) & 7, s = idx >> 12;
    int n = lane & 15, hf = lane >> 4;
    w1f[s][t][lane][i] = (_Float16)W1[(128 + s * 32 + hf * 8 + (i < 8 ? i : i + 8)) * 128 + t * 16 + n];
  }
  for (int idx = tid; idx < 4 * 4 * 32 * 16; idx += 128) {
    int i = idx & 15, lane = (idx >> 4) & 31, u = (idx >> 9) & 3, s2 = idx >> 11;
    int n = lane & 15, hf = lane >> 4;
    w2f[s2][u][lane][i] = (_Float16)W2[(s2 * 32 + hf * 8 + (i < 8 ? i : i + 8)) * 64 + u * 16 + n];
  }

  const int wave = tid >> 5;
  const int lane = tid & 31;
  const int m    = lane & 15;       // row (k) / col index within 16
  const int hf   = lane >> 4;       // half-wave
  const int k0   = kbase + wave * 16;
  const int krow = wave * 16 + m;

  for (int qi = 0; qi < 16; ++qi) {
    const int q = qbase + qi;
    __syncthreads();                       // WAR on qe_s/qh_s from previous iter
    for (int idx = tid; idx < 192; idx += 128) {
      int d = idx & 63, c = idx >> 6;
      qe_s[c][d] = q_equi[(((b * NQ + q) * 3 + c) << 6) + d];
    }
    qh_s[tid] = qh[(b * NQ + q) * 128 + tid];
    __syncthreads();

    // ---- 1. dotprod A-fragments: lane m holds row m, K(i,hf)=(i<8?i:i+8)+8*hf
    v16h a1[2];
#pragma unroll
    for (int frag = 0; frag < 2; ++frag)
#pragma unroll
      for (int ip = 0; ip < 8; ++ip) {
        int i = ip * 2;
        int d = frag * 32 + (i < 8 ? i : i + 8) + hf * 8;   // d, d+1 pair
        float2 acc = {0.f, 0.f};
#pragma unroll
        for (int c = 0; c < 3; ++c) {
          float2 qv = *(const float2*)&qe_s[c][d];
          float2 kv = *(const float2*)&ke_s[c][krow][d];
          acc.x += qv.x * kv.x;
          acc.y += qv.y * kv.y;
        }
        a1[frag][i]     = (_Float16)acc.x;
        a1[frag][i + 1] = (_Float16)acc.y;
      }

    // ---- 2. GEMM1: H_pre = D @ W1c (K=64 -> 2 steps, N=128 -> 8 tiles) ----
    v8f hacc[8];
#pragma unroll
    for (int t = 0; t < 8; ++t) {
      v8f c = {};
#pragma unroll
      for (int s = 0; s < 2; ++s) {
        v16h bf = *(const v16h*)&w1f[s][t][lane][0];
        c = __builtin_amdgcn_wmma_f32_16x16x32_f16(false, a1[s], false, bf,
                                                   (short)0, c, false, false);
      }
      hacc[t] = c;
    }

    // ---- 3. epilogue: + qh + kh(LDS), SiLU, write A2-fragment-packed f16 ----
    // C/D layout: c[r] = C[M = r + 8*hf, N = m];  value (Mrow, f) -> packed slot
#pragma unroll
    for (int t = 0; t < 8; ++t)
#pragma unroll
      for (int r = 0; r < 8; ++r) {
        int Mrow = r + 8 * hf;
        int f    = t * 16 + m;
        float x = hacc[t][r] + qh_s[f] + kh_s[(k0 - kbase) + Mrow - wave * 16 + wave * 16][f];
        x = x / (1.f + __expf(-x));            // SiLU
        int s2 = f >> 5;
        int r32 = f & 31;
        int dhf = (r32 >> 3) & 1;
        int di  = (r32 & 7) | ((r32 & 16) >> 1);
        h_pack[wave][s2][dhf * 16 + Mrow][di] = (_Float16)x;
      }

    // intra-wave LDS RAW (cross-lane): LDS is in-order per wave; force wait and
    // stop the compiler hoisting the packed reloads above the stores.
    asm volatile("s_wait_dscnt 0" ::: "memory");

    // ---- 4. A2 fragments: one b128-pair load per fragment ----
    v16h a2[4];
#pragma unroll
    for (int s2 = 0; s2 < 4; ++s2)
      a2[s2] = *(const v16h*)&h_pack[wave][s2][lane][0];

    // ---- 5. GEMM2: O = H @ W2 (K=128 -> 4 steps, N=64 -> 4 tiles) + store ----
#pragma unroll
    for (int u = 0; u < 4; ++u) {
      v8f c = {};
#pragma unroll
      for (int s2 = 0; s2 < 4; ++s2) {
        v16h bf = *(const v16h*)&w2f[s2][u][lane][0];
        c = __builtin_amdgcn_wmma_f32_16x16x32_f16(false, a2[s2], false, bf,
                                                   (short)0, c, false, false);
      }
#pragma unroll
      for (int r = 0; r < 8; ++r) {
        int Mrow = r + 8 * hf;
        int o    = u * 16 + m;
        out[(size_t)((b * NQ + q) * NK + k0 + Mrow) * 64 + o] = c[r] + b2_s[o];
      }
    }
  }
}

// ---------------------------------------------------------------------------
extern "C" void kernel_launch(void* const* d_in, const int* in_sizes, int n_in,
                              void* d_out, int out_size, void* d_ws, size_t ws_size,
                              hipStream_t stream)
{
  const float* q_equi = (const float*)d_in[0];
  const float* q_inv  = (const float*)d_in[1];
  const float* k_equi = (const float*)d_in[2];
  const float* k_inv  = (const float*)d_in[3];
  const float* Wq     = (const float*)d_in[4];
  const float* bq     = (const float*)d_in[5];
  const float* Wk     = (const float*)d_in[6];
  const float* bk     = (const float*)d_in[7];
  const float* W1     = (const float*)d_in[8];
  const float* b1     = (const float*)d_in[9];
  const float* W2     = (const float*)d_in[10];
  const float* b2     = (const float*)d_in[11];
  float* out = (float*)d_out;

  float* qh = (float*)d_ws;                 // [B*NQ, 128] f32
  float* kh = qh + BB * NQ * 128;           // [B*NK, 128] f32  (1.5 MB total)

  precompute_qh_kh<<<2 * BB * NQ, 128, 0, stream>>>(q_inv, k_inv, Wq, bq, Wk, bk,
                                                    W1, b1, qh, kh);
  fused_pairwise<<<BB * (NQ / 16) * 6, 128, 0, stream>>>(q_equi, k_equi, qh, kh,
                                                         W1, W2, b2, out);
}
